// AddLinearAttention_40381282517340
// MI455X (gfx1250) — matrix operand order repaired and checked by reading, not compiled, not run on hardware
//
#include <hip/hip_runtime.h>
#include <hip/hip_bf16.h>

// ---------------------------------------------------------------------------
// AddLinearAttention for MI455X (gfx1250), wave32 + WMMA f16 (f32 accumulate).
//
// All GEMM operands are pre-converted to f16 in fragment-friendly layouts
// (A: M x K row-major, B given as B^T: N x K row-major, K contiguous), so the
// GEMM loads fragments straight from global memory with global_load_b128.
// Every f16 operand (<=38MB) is L2-resident (192MB L2), so cross-block
// re-reads are L2 hits and HBM traffic is ~compulsory (~0.5GB @ 23.3TB/s).
//
// The GEMM K-loop is software-pipelined: iteration k's WMMAs consume
// fragments loaded at k-1 while k+1's loads are in flight, so the waits in
// front of the WMMA chain are partial (latency hidden by matrix ops).
//
// The x transpose kernel exercises the CDNA5 async-to-LDS path:
// global_load_async_to_lds_b128 + s_wait_asynccnt (CDNA5 ISA).
// ---------------------------------------------------------------------------

#define BATCH 16
#define CC    128
#define HH    96
#define WW    96
#define HWP   (HH * WW)                 // 9216
#define SCALEC 0.08838834764831845f     // C^-0.5

typedef __attribute__((ext_vector_type(16))) _Float16 v16h;
typedef __attribute__((ext_vector_type(8)))  _Float16 v8h;
typedef __attribute__((ext_vector_type(8)))  float    v8f;

// ---------------------------------------------------------------------------
// WMMA GEMM, fragments direct from global (L2-resident f16 operands):
//   Out[m,n] = outScale * sum_k A[m*lda+k] * Bt[n*ldb+k]  (+ bias[m])
// Block: 256 thr = 8 waves (2M x 4N); wave tile 16 x (16*WNSUB); K step 32.
// OUT_F16T: store f16 at Out[n*ldo+m] (transposed) instead of f32 [m*ldo+n].
// ---------------------------------------------------------------------------
template <int WNSUB, bool OUT_F16T>
__global__ __launch_bounds__(256) void wmma_gemm_kernel(
    const _Float16* __restrict__ A, const _Float16* __restrict__ Bt,
    const float* __restrict__ bias, void* __restrict__ Out,
    int K, int lda, int ldb, int ldo,
    long long sA, long long sB, long long sO, float outScale)
{
    const int tid     = threadIdx.x;
    const int lane    = tid & 31;
    const int wid     = tid >> 5;
    const int wm      = wid & 1;
    const int wn      = wid >> 1;
    const int half_id = lane >> 4;
    const int r       = lane & 15;

    const int m0 = blockIdx.y * 32;
    const int n0 = blockIdx.x * (64 * WNSUB);

    const _Float16* Ab  = A  + (size_t)sA * blockIdx.z;
    const _Float16* Btb = Bt + (size_t)sB * blockIdx.z;

    // per-lane fragment pointers (K-contiguous rows)
    const _Float16* Ap = Ab + (size_t)(m0 + wm * 16 + r) * lda;
    const _Float16* Bp[WNSUB];
#pragma unroll
    for (int s = 0; s < WNSUB; ++s)
        Bp[s] = Btb + (size_t)(n0 + wn * (16 * WNSUB) + s * 16 + r) * ldb;

    v8f acc[WNSUB];
#pragma unroll
    for (int s = 0; s < WNSUB; ++s)
#pragma unroll
        for (int j = 0; j < 8; ++j) acc[s][j] = 0.0f;

    // ---- software pipeline: preload k-step 0 ----
    // A fragment 16x32: lane half h holds K = h*16 .. h*16+15
    // B fragment 32x16: lane half h holds K = h*16 .. h*16+15
    v8h alo = *reinterpret_cast<const v8h*>(Ap + half_id * 16);
    v8h ahi = *reinterpret_cast<const v8h*>(Ap + half_id * 16 + 8);
    v8h blo[WNSUB], bhi[WNSUB];
#pragma unroll
    for (int s = 0; s < WNSUB; ++s) {
        blo[s] = *reinterpret_cast<const v8h*>(Bp[s] + half_id * 16);
        bhi[s] = *reinterpret_cast<const v8h*>(Bp[s] + half_id * 16 + 8);
    }

    for (int kb = 0; kb < K - 32; kb += 32) {
        // issue NEXT k-step loads first (overlap with this step's WMMAs)
        const int kn = kb + 32;
        v8h alo_n = *reinterpret_cast<const v8h*>(Ap + kn + half_id * 16);
        v8h ahi_n = *reinterpret_cast<const v8h*>(Ap + kn + half_id * 16 + 8);
        v8h blo_n[WNSUB], bhi_n[WNSUB];
#pragma unroll
        for (int s = 0; s < WNSUB; ++s) {
            blo_n[s] = *reinterpret_cast<const v8h*>(Bp[s] + kn + half_id * 16);
            bhi_n[s] = *reinterpret_cast<const v8h*>(Bp[s] + kn + half_id * 16 + 8);
        }
        __builtin_prefetch(Bp[0] + kb + 64, 0, 1);  // speculative, fault-safe

        v16h afrag;
#pragma unroll
        for (int i = 0; i < 8; ++i) { afrag[i] = alo[i]; afrag[i + 8] = ahi[i]; }
#pragma unroll
        for (int s = 0; s < WNSUB; ++s) {
            v16h bfrag;
#pragma unroll
            for (int i = 0; i < 8; ++i) { bfrag[i] = blo[s][i]; bfrag[i + 8] = bhi[s][i]; }
            acc[s] = __builtin_amdgcn_wmma_f32_16x16x32_f16(
                false, afrag, false, bfrag, (short)0, acc[s], false, false);
        }
        // rotate pipeline registers
        alo = alo_n; ahi = ahi_n;
#pragma unroll
        for (int s = 0; s < WNSUB; ++s) { blo[s] = blo_n[s]; bhi[s] = bhi_n[s]; }
    }

    // ---- pipeline epilogue: final k-step (no loads) ----
    {
        v16h afrag;
#pragma unroll
        for (int i = 0; i < 8; ++i) { afrag[i] = alo[i]; afrag[i + 8] = ahi[i]; }
#pragma unroll
        for (int s = 0; s < WNSUB; ++s) {
            v16h bfrag;
#pragma unroll
            for (int i = 0; i < 8; ++i) { bfrag[i] = blo[s][i]; bfrag[i + 8] = bhi[s][i]; }
            acc[s] = __builtin_amdgcn_wmma_f32_16x16x32_f16(
                false, afrag, false, bfrag, (short)0, acc[s], false, false);
        }
    }

    // C/D layout: VGPR j -> M = j + 8*half_id ; lane%16 -> N
#pragma unroll
    for (int s = 0; s < WNSUB; ++s) {
        const int col = n0 + wn * (16 * WNSUB) + s * 16 + r;
#pragma unroll
        for (int j = 0; j < 8; ++j) {
            const int m = m0 + wm * 16 + half_id * 8 + j;
            float v = acc[s][j] * outScale;
            if (bias) v += bias[m];
            if (OUT_F16T) {
                _Float16* O = (_Float16*)Out + (size_t)sO * blockIdx.z;
                O[(size_t)col * ldo + m] = (_Float16)v;
            } else {
                float* O = (float*)Out + (size_t)sO * blockIdx.z;
                O[(size_t)m * ldo + col] = v;
            }
        }
    }
}

// ---------------------------------------------------------------------------
// x (B,C,HW) f32 -> x_t (B,HW,C) f16, 64x64 tiles.
// Tile fill uses CDNA5 async global->LDS copies (ASYNCcnt-tracked).
// ---------------------------------------------------------------------------
__global__ __launch_bounds__(256) void transpose_cvt_kernel(
    const float* __restrict__ in, _Float16* __restrict__ out)
{
    __shared__ __align__(16) float Tile[64][72];   // pad 8 floats vs bank stride
    const int p0  = blockIdx.x * 64;
    const int c0  = blockIdx.y * 64;
    const int b   = blockIdx.z;
    const int tid = threadIdx.x;

    const unsigned ldsBase = (unsigned)(unsigned long long)(&Tile[0][0]);
    // phase 1: 64 rows x 256B, 16B per lane per issue, via async-to-LDS
#pragma unroll
    for (int i = 0; i < 4; ++i) {
        const int e = tid + i * 256;
        const int row = e >> 4, chunk = e & 15;
        const float* g = in + ((size_t)b * CC + c0 + row) * HWP + p0 + chunk * 4;
        const unsigned l = ldsBase + (unsigned)(row * 72 + chunk * 4) * 4u;
        asm volatile("global_load_async_to_lds_b128 %0, %1, off"
                     :: "v"(l), "v"((unsigned long long)g) : "memory");
    }
#if __has_builtin(__builtin_amdgcn_s_wait_asynccnt)
    __builtin_amdgcn_s_wait_asynccnt(0);
#else
    asm volatile("s_wait_asynccnt 0x0" ::: "memory");
#endif
    __syncthreads();

    // phase 2: transposed read, f16 store coalesced over c
    const int cc = tid & 63;
#pragma unroll
    for (int i = 0; i < 16; ++i) {
        const int pp = (tid >> 6) + i * 4;
        out[((size_t)b * HWP + p0 + pp) * CC + c0 + cc] = (_Float16)Tile[cc][pp];
    }
}

// flat f32 -> f16 convert (weights)
__global__ __launch_bounds__(256) void cvt_f16_kernel(
    const float* __restrict__ in, _Float16* __restrict__ out, int n)
{
    const int i = blockIdx.x * 256 + threadIdx.x;
    if (i < n) out[i] = (_Float16)in[i];
}

// ---------------------------------------------------------------------------
// Depthwise conv (3x3/5x5) on a channel slice of qkvo (B,4C,HW) f32,
// optional residual add + elu()+1, f16 outputs in token and/or channel major.
// ---------------------------------------------------------------------------
__global__ __launch_bounds__(256) void dwconv_kernel(
    const float* __restrict__ in, int chanOff,
    const float* __restrict__ w, const float* __restrict__ bias,
    _Float16* __restrict__ outTok, _Float16* __restrict__ outCh,
    int ksize, int pad, int addResidual, int applyElu)
{
    const size_t idx = (size_t)blockIdx.x * 256 + threadIdx.x;
    const size_t total = (size_t)BATCH * CC * HWP;
    if (idx >= total) return;
    const int p = (int)(idx % HWP);
    const int c = (int)((idx / HWP) % CC);
    const int b = (int)(idx / ((size_t)HWP * CC));
    const int y = p / WW, x = p % WW;

    const size_t base = ((size_t)b * (4 * CC) + chanOff + c) * HWP;
    float acc = bias[c];
    const float* wc = w + c * ksize * ksize;
    for (int ky = 0; ky < ksize; ++ky) {
        const int iy = y + ky - pad;
        if (iy < 0 || iy >= HH) continue;
        for (int kx = 0; kx < ksize; ++kx) {
            const int ix = x + kx - pad;
            if (ix < 0 || ix >= WW) continue;
            acc += wc[ky * ksize + kx] * in[base + (size_t)iy * WW + ix];
        }
    }
    float val = addResidual ? in[base + p] + acc : acc;
    if (applyElu) val = (val > 0.0f ? val : __expf(val) - 1.0f) + 1.0f;
    const _Float16 hv = (_Float16)val;
    if (outTok) outTok[((size_t)b * HWP + p) * CC + c] = hv;
    if (outCh)  outCh[((size_t)b * CC + c) * HWP + p] = hv;
}

// kmean[row] = mean over HW of channel-major f16 row (row = b*C + c)
__global__ __launch_bounds__(256) void rowmean_kernel(
    const _Float16* __restrict__ in, float* __restrict__ out)
{
    __shared__ float red[256];
    const size_t row = blockIdx.x;
    float s = 0.0f;
    for (int i = threadIdx.x; i < HWP; i += 256)
        s += (float)in[row * HWP + i];
    red[threadIdx.x] = s;
    __syncthreads();
    for (int st = 128; st > 0; st >>= 1) {
        if (threadIdx.x < st) red[threadIdx.x] += red[threadIdx.x + st];
        __syncthreads();
    }
    if (threadIdx.x == 0) out[row] = red[0] * (1.0f / HWP);
}

// Per-token epilogue: z = scale*<q,kmean>; vmean = mean_d v;
// t[b,n,d] = (rraw*(1+1/(z+1e-6)) - z*vmean + lepe) * o   (f16, token-major)
__global__ __launch_bounds__(CC) void epilogue_kernel(
    const _Float16* __restrict__ q_tok, const _Float16* __restrict__ v_tok,
    const float* __restrict__ resraw, const float* __restrict__ kmean,
    const _Float16* __restrict__ lepe_tok, const float* __restrict__ qkvo,
    _Float16* __restrict__ t_tok)
{
    const int tix = blockIdx.x;          // b*HW + n
    const int d   = threadIdx.x;
    const int b   = tix / HWP;
    const int n   = tix % HWP;
    __shared__ float sz[CC], sv[CC];
    const float qv = (float)q_tok[(size_t)tix * CC + d];
    const float vv = (float)v_tok[(size_t)tix * CC + d];
    sz[d] = qv * kmean[b * CC + d];
    sv[d] = vv;
    __syncthreads();
    for (int s = CC / 2; s > 0; s >>= 1) {
        if (d < s) { sz[d] += sz[d + s]; sv[d] += sv[d + s]; }
        __syncthreads();
    }
    const float z     = sz[0] * SCALEC;
    const float vmean = sv[0] * (1.0f / CC);
    const float f     = 1.0f + 1.0f / (z + 1e-6f);
    const float rr    = resraw[(size_t)tix * CC + d] * f - z * vmean;
    const float lp    = (float)lepe_tok[(size_t)tix * CC + d];
    const float ov    = qkvo[((size_t)b * 4 * CC + 3 * CC + d) * HWP + n];
    t_tok[(size_t)tix * CC + d] = (_Float16)((rr + lp) * ov);
}

extern "C" void kernel_launch(void* const* d_in, const int* in_sizes, int n_in,
                              void* d_out, int out_size, void* d_ws, size_t ws_size,
                              hipStream_t stream)
{
    (void)in_sizes; (void)n_in; (void)out_size; (void)ws_size;
    const float* x      = (const float*)d_in[0];
    // d_in[1]=sin, d_in[2]=cos : unused by the reference
    const float* w_qkvo = (const float*)d_in[3];
    const float* b_qkvo = (const float*)d_in[4];
    const float* w_lepe = (const float*)d_in[5];
    const float* b_lepe = (const float*)d_in[6];
    const float* w_proj = (const float*)d_in[7];
    const float* b_proj = (const float*)d_in[8];
    const float* w_q    = (const float*)d_in[9];
    const float* b_q    = (const float*)d_in[10];
    const float* w_k    = (const float*)d_in[11];
    const float* b_k    = (const float*)d_in[12];
    const float* w_v    = (const float*)d_in[13];
    const float* b_v    = (const float*)d_in[14];
    float* out = (float*)d_out;

    // ---- workspace layout ----
    const size_t nQKVO = (size_t)BATCH * 4 * CC * HWP;   // f32
    const size_t nTOK  = (size_t)BATCH * HWP * CC;       // per tensor
    float* wsf    = (float*)d_ws;
    float* qkvo   = wsf;                 wsf += nQKVO;
    float* resraw = wsf;                 wsf += nTOK;
    float* kmean  = wsf;                 wsf += (size_t)BATCH * CC;
    _Float16* wsh     = (_Float16*)wsf;
    _Float16* x_t_h   = wsh;             wsh += nTOK;
    _Float16* q_tok_h = wsh;             wsh += nTOK;
    _Float16* v_tok_h = wsh;             wsh += nTOK;
    _Float16* k_ch_h  = wsh;             wsh += nTOK;
    _Float16* v_ch_h  = wsh;             wsh += nTOK;
    _Float16* lepe_h  = wsh;             wsh += nTOK;
    _Float16* t_tok_h = wsh;             wsh += nTOK;
    _Float16* kvT_h   = wsh;             wsh += (size_t)BATCH * CC * CC;
    _Float16* w_qkvo_h = wsh;            wsh += (size_t)4 * CC * CC;
    _Float16* w_proj_h = wsh;            wsh += (size_t)CC * CC;

    const long long llTOK = (long long)HWP * CC;
    const long long llKV  = (long long)CC * CC;
    const float s2 = SCALEC / (float)HWP;   // ((scale/hw)^0.5)^2 folded into kv

    // 0. weight converts + x transpose (async-LDS path)
    cvt_f16_kernel<<<(4 * CC * CC + 255) / 256, 256, 0, stream>>>(w_qkvo, w_qkvo_h, 4 * CC * CC);
    cvt_f16_kernel<<<(CC * CC + 255) / 256, 256, 0, stream>>>(w_proj, w_proj_h, CC * CC);
    transpose_cvt_kernel<<<dim3(HWP / 64, CC / 64, BATCH), 256, 0, stream>>>(x, x_t_h);

    // 1. qkvo = W_qkvo @ X  (M=512, N=9216, K=128)
    wmma_gemm_kernel<4, false><<<dim3(HWP / 256, (4 * CC) / 32, BATCH), 256, 0, stream>>>(
        w_qkvo_h, x_t_h, b_qkvo, qkvo, CC, CC, CC, HWP,
        0LL, llTOK, (long long)4 * CC * HWP, 1.0f);

    const int dwBlocks = (int)(((size_t)BATCH * CC * HWP + 255) / 256);
    // 2-3. depthwise stages
    dwconv_kernel<<<dwBlocks, 256, 0, stream>>>(qkvo, 2 * CC, w_lepe, b_lepe,
                                                lepe_h, nullptr, 5, 2, 0, 0);
    dwconv_kernel<<<dwBlocks, 256, 0, stream>>>(qkvo, 0,      w_q, b_q,
                                                q_tok_h, nullptr, 3, 1, 1, 1);
    dwconv_kernel<<<dwBlocks, 256, 0, stream>>>(qkvo, CC,     w_k, b_k,
                                                nullptr, k_ch_h, 3, 1, 1, 1);
    dwconv_kernel<<<dwBlocks, 256, 0, stream>>>(qkvo, 2 * CC, w_v, b_v,
                                                v_tok_h, v_ch_h, 3, 1, 1, 0);

    // 4. kmean
    rowmean_kernel<<<BATCH * CC, 256, 0, stream>>>(k_ch_h, kmean);

    // 5. kvT = s^2 * (k^T v)^T  (M=N=128, K=9216; f16 transposed out)
    wmma_gemm_kernel<2, true><<<dim3(1, CC / 32, BATCH), 256, 0, stream>>>(
        k_ch_h, v_ch_h, nullptr, kvT_h, HWP, HWP, HWP, CC,
        llTOK, llTOK, llKV, s2);

    // 6. resraw = q_tok @ kv  (M=9216, N=128, K=128)
    wmma_gemm_kernel<2, false><<<dim3(1, HWP / 32, BATCH), 256, 0, stream>>>(
        q_tok_h, kvT_h, nullptr, resraw, CC, CC, CC, CC,
        llTOK, llKV, llTOK, 1.0f);

    // 7. epilogue -> t (token-major f16 == t^T for proj GEMM)
    epilogue_kernel<<<BATCH * HWP, CC, 0, stream>>>(q_tok_h, v_tok_h, resraw,
                                                    kmean, lepe_h, qkvo, t_tok_h);

    // 8. out = W_proj @ t + b_proj  (M=128, N=9216, K=128)
    wmma_gemm_kernel<4, false><<<dim3(HWP / 256, CC / 32, BATCH), 256, 0, stream>>>(
        w_proj_h, t_tok_h, b_proj, out, CC, CC, CC, HWP,
        0LL, llTOK, llTOK, 1.0f);
}
